// TransformerModule_50929722196280
// MI455X (gfx1250) — matrix operand order repaired and checked by reading, not compiled, not run on hardware
//
#include <hip/hip_runtime.h>
#include <hip/hip_bf16.h>
#include <math.h>

// ---------------------------------------------------------------------------
// Transformer forward for MI455X (gfx1250, wave32, WMMA + TDM).
// GEMMs: v_wmma_f32_16x16x32_bf16, weights DMA'd to LDS via tensor_load_to_lds.
// ---------------------------------------------------------------------------

#define DIM      512
#define VOCAB    32000
#define HEADS    8
#define HEAD_DIM 64
#define LAYERS   6
#define SEQ      512
#define BATCH    8
#define FFDIM    2048
#define TOK      (BATCH * SEQ)
#define LN_EPS   1e-5f

typedef __bf16 bf16_t;
typedef __attribute__((ext_vector_type(2)))  __bf16 bf16x2;
typedef __attribute__((ext_vector_type(4)))  __bf16 bf16x4;
typedef __attribute__((ext_vector_type(16))) __bf16 v16bf;
typedef __attribute__((ext_vector_type(8)))  float  v8f;
typedef __attribute__((ext_vector_type(4)))  unsigned uint4v;
typedef __attribute__((ext_vector_type(8)))  unsigned uint8v;

// ---- WMMA fragment helpers (layouts: CDNA5 ISA §7.12.2) --------
// A (16x32 bf16), LDS row-major [m][k] stride 32: lane L: M=L&15, kh=L>>4;
// element i = K kh*16+i (contiguous 16 bf16).
__device__ __forceinline__ v16bf frag_a(const bf16_t* base, int lane) {
  int m = lane & 15, kh = lane >> 4;
  const bf16_t* p = base + m * 32 + kh * 16;
  v16bf f;
#pragma unroll
  for (int i = 0; i < 16; ++i) f[i] = p[i];
  return f;
}

// B (32x16 bf16) from TRANSPOSED LDS [n][k] stride 32 (contiguous 16 bf16).
__device__ __forceinline__ v16bf frag_b_tr(const bf16_t* baseT, int lane) {
  int n = lane & 15, kh = lane >> 4;
  const bf16_t* p = baseT + n * 32 + kh * 16;
  v16bf f;
#pragma unroll
  for (int i = 0; i < 16; ++i) f[i] = p[i];
  return f;
}

// B (32x16 bf16) from ROW-MAJOR LDS [k][n] stride 128 (TDM's natural layout).
__device__ __forceinline__ v16bf frag_b_rm(const bf16_t* base, int n_off, int lane) {
  int n = n_off + (lane & 15), kh = lane >> 4;
  const bf16_t* p = base + (kh * 16) * 128 + n;
  v16bf f;
#pragma unroll
  for (int i = 0; i < 16; ++i) f[i] = p[i * 128];
  return f;
}

__device__ __forceinline__ v8f wmma_bf16(v16bf a, v16bf b, v8f c) {
  return __builtin_amdgcn_wmma_f32_16x16x32_bf16(false, a, false, b, (short)0, c,
                                                 false, false);
}

__device__ __forceinline__ bf16x4 pack4(float x, float y, float z, float w) {
  bf16x4 o; o[0] = (bf16_t)x; o[1] = (bf16_t)y; o[2] = (bf16_t)z; o[3] = (bf16_t)w;
  return o;
}

// ---------------------------------------------------------------------------
// f32 -> bf16 conversion pass (weights).  4 elems/thread, b128 in / b64 out.
// ---------------------------------------------------------------------------
__global__ __launch_bounds__(256) void cvt_bf16_kernel(const float* __restrict__ src,
                                                       bf16_t* __restrict__ dst) {
  size_t i = ((size_t)blockIdx.x * 256 + threadIdx.x) * 4;
  float4 f = *(const float4*)(src + i);
  *(bf16x4*)(dst + i) = pack4(f.x, f.y, f.z, f.w);
}

// ---------------------------------------------------------------------------
// GEMM: C[M,N] = act(A[M,K]_f32 @ W[K,N]_bf16 + bias).
// Block tile 128x128, K-step 32, 256 threads = 8 waves x (64x32) sub-tiles.
// A tile: vectorized f32->bf16 staging.  W tile: TDM tensor_load_to_lds,
// issued by wave 0, completed with s_wait_tensorcnt (CDNA5 ISA-8).
// ---------------------------------------------------------------------------
template <int BIAS, int RELU>
__global__ __launch_bounds__(256) void gemm_kernel(const float* __restrict__ A,
                                                   const bf16_t* __restrict__ Bw,
                                                   const float* __restrict__ bias,
                                                   float* __restrict__ C,
                                                   int M, int N, int K) {
  __shared__ bf16_t As[128 * 32];  // [m][k]
  __shared__ bf16_t Bs[32 * 128];  // [k][n]  (TDM destination, row-major)
  const int tid  = threadIdx.x;
  const int lane = tid & 31;
  const int w    = tid >> 5;
  const int wm   = (w >> 2) * 64;
  const int wn   = (w & 3) * 32;
  const int m0   = blockIdx.y * 128;
  const int n0   = blockIdx.x * 128;

  v8f z = {0.f, 0.f, 0.f, 0.f, 0.f, 0.f, 0.f, 0.f};
  v8f acc[4][2];
#pragma unroll
  for (int i = 0; i < 4; ++i)
#pragma unroll
    for (int j = 0; j < 2; ++j) acc[i][j] = z;

  // Invariant pieces of the tensor descriptor (wave-uniform).
  const unsigned lds_b = (unsigned)(uintptr_t)&Bs[0];
  const unsigned long long gbase = (unsigned long long)(uintptr_t)Bw;

  const int kt_end = K >> 5;
  for (int kt = 0; kt < kt_end; ++kt) {
    const int kb = kt << 5;
    // ---- A tile: 128x32 f32 -> bf16, float4 loads + b64 LDS stores ----
#pragma unroll
    for (int r = 0; r < 4; ++r) {
      int l4 = tid + r * 256;          // 1024 float4 per tile
      int row = l4 >> 3, kq = (l4 & 7) * 4;
      float4 f = *(const float4*)&A[(size_t)(m0 + row) * K + kb + kq];
      *(bf16x4*)&As[row * 32 + kq] = pack4(f.x, f.y, f.z, f.w);
    }
    if (kt + 1 < kt_end)
      __builtin_prefetch(&A[(size_t)(m0 + (tid >> 1)) * K + kb + 32], 0, 1);

    // ---- W tile: TDM DMA of 32x128 bf16 into LDS (wave 0 only) ----
    if (w == 0) {
      unsigned long long ga = gbase + 2ull * ((unsigned long long)kb * (unsigned)N + (unsigned)n0);
      uint4v g0;
      g0[0] = 1u;                                   // count=1, user mode
      g0[1] = lds_b;                                // lds_addr
      g0[2] = (unsigned)(ga & 0xffffffffull);       // global_addr[31:0]
      g0[3] = (unsigned)((ga >> 32) & 0x01ffffffull) | (2u << 30);  // [56:32] | type=2
      uint8v g1;
      g1[0] = 1u << 16;                             // data_size=1 (2 bytes)
      g1[1] = ((unsigned)N & 0xffffu) << 16;        // tensor_dim0[15:0]
      g1[2] = (((unsigned)N >> 16) & 0xffffu) | (((unsigned)K & 0xffffu) << 16);
      g1[3] = (((unsigned)K >> 16) & 0xffffu) | (128u << 16);   // tile_dim0=128
      g1[4] = 32u;                                  // tile_dim1=32, tile_dim2=0
      g1[5] = (unsigned)N;                          // tensor_dim0_stride[31:0]
      g1[6] = 0u;
      g1[7] = 0u;
      asm volatile("tensor_load_to_lds %0, %1" :: "s"(g0), "s"(g1) : "memory");
      __builtin_amdgcn_s_wait_tensorcnt(0);
    }
    __syncthreads();

    v16bf af[4];
#pragma unroll
    for (int mt = 0; mt < 4; ++mt) af[mt] = frag_a(&As[(wm + mt * 16) * 32], lane);
    v16bf bf0 = frag_b_rm(Bs, wn, lane);
    v16bf bf1 = frag_b_rm(Bs, wn + 16, lane);
#pragma unroll
    for (int mt = 0; mt < 4; ++mt) {
      acc[mt][0] = wmma_bf16(af[mt], bf0, acc[mt][0]);
      acc[mt][1] = wmma_bf16(af[mt], bf1, acc[mt][1]);
    }
    __syncthreads();
  }

  const int nlo = lane & 15, mh = lane >> 4;
#pragma unroll
  for (int mt = 0; mt < 4; ++mt)
#pragma unroll
    for (int nt = 0; nt < 2; ++nt) {
      int n = n0 + wn + nt * 16 + nlo;
      float bv = BIAS ? bias[n] : 0.f;
#pragma unroll
      for (int r = 0; r < 8; ++r) {
        int m = m0 + wm + mt * 16 + mh * 8 + r;
        float v = acc[mt][nt][r] + bv;
        if (RELU) v = v > 0.f ? v : 0.f;
        C[(size_t)m * N + n] = v;
      }
    }
}

// ---------------------------------------------------------------------------
// Attention scores: S[bh,q,k] = (Q_h . K_h) / 8.   64x64 tile, K = 2x32.
// ---------------------------------------------------------------------------
__global__ __launch_bounds__(256) void attn_scores_kernel(const float* __restrict__ Q,
                                                          const float* __restrict__ Km,
                                                          float* __restrict__ Sout) {
  __shared__ bf16_t Qs[64 * 32];
  __shared__ bf16_t KsT[64 * 32];
  const int tid = threadIdx.x, lane = tid & 31, w = tid >> 5;
  const int wq = (w >> 1) * 16, wn = (w & 1) * 32;
  const int q0 = blockIdx.x * 64;
  const int k0 = blockIdx.y * 64;
  const int bh = blockIdx.z;
  const int b = bh / HEADS, h = bh % HEADS;
  const size_t rowbase = (size_t)b * SEQ;

  v8f acc0 = {0.f, 0.f, 0.f, 0.f, 0.f, 0.f, 0.f, 0.f};
  v8f acc1 = acc0;

  for (int kt = 0; kt < HEAD_DIM / 32; ++kt) {
    const int kd0 = kt * 32;
#pragma unroll
    for (int r = 0; r < 2; ++r) {
      int l4 = tid + r * 256;          // 512 float4 per tile
      int row = l4 >> 3, kd = (l4 & 7) * 4;
      float4 fq = *(const float4*)&Q[(rowbase + q0 + row) * DIM + h * HEAD_DIM + kd0 + kd];
      float4 fk = *(const float4*)&Km[(rowbase + k0 + row) * DIM + h * HEAD_DIM + kd0 + kd];
      *(bf16x4*)&Qs[row * 32 + kd]  = pack4(fq.x, fq.y, fq.z, fq.w);
      *(bf16x4*)&KsT[row * 32 + kd] = pack4(fk.x, fk.y, fk.z, fk.w);
    }
    __syncthreads();
    v16bf aq = frag_a(&Qs[wq * 32], lane);
    v16bf b0 = frag_b_tr(&KsT[(wn) * 32], lane);
    v16bf b1 = frag_b_tr(&KsT[(wn + 16) * 32], lane);
    acc0 = wmma_bf16(aq, b0, acc0);
    acc1 = wmma_bf16(aq, b1, acc1);
    __syncthreads();
  }
  const int nlo = lane & 15, mh = lane >> 4;
  const float scale = 0.125f;  // 1/sqrt(64)
  const size_t sb = (size_t)bh * SEQ * SEQ;
#pragma unroll
  for (int r = 0; r < 8; ++r) {
    int q = q0 + wq + mh * 8 + r;
    Sout[sb + (size_t)q * SEQ + k0 + wn + nlo]      = acc0[r] * scale;
    Sout[sb + (size_t)q * SEQ + k0 + wn + 16 + nlo] = acc1[r] * scale;
  }
}

// ---------------------------------------------------------------------------
// Masked softmax over key axis, in place.  One block per (bh,q) row.
// ---------------------------------------------------------------------------
__global__ __launch_bounds__(256) void softmax_kernel(float* __restrict__ Sio,
                                                      const int* __restrict__ key_tok,
                                                      int causal) {
  __shared__ float red[256];
  const int row = blockIdx.x;
  const int q   = row & (SEQ - 1);
  const int bh  = row >> 9;
  const int b   = bh / HEADS;
  const int tid = threadIdx.x;
  float v[2];
  float mx = -1e30f;
#pragma unroll
  for (int j = 0; j < 2; ++j) {
    int c = tid + j * 256;
    bool valid = (key_tok[b * SEQ + c] != 0) && (!causal || c <= q);
    float s = Sio[(size_t)row * SEQ + c];
    v[j] = valid ? s : -1e9f;
    mx = fmaxf(mx, v[j]);
  }
  red[tid] = mx;
  __syncthreads();
  for (int off = 128; off > 0; off >>= 1) {
    if (tid < off) red[tid] = fmaxf(red[tid], red[tid + off]);
    __syncthreads();
  }
  mx = red[0];
  __syncthreads();
  float sum = 0.f;
#pragma unroll
  for (int j = 0; j < 2; ++j) { v[j] = __expf(v[j] - mx); sum += v[j]; }
  red[tid] = sum;
  __syncthreads();
  for (int off = 128; off > 0; off >>= 1) {
    if (tid < off) red[tid] += red[tid + off];
    __syncthreads();
  }
  float inv = 1.f / red[0];
#pragma unroll
  for (int j = 0; j < 2; ++j) Sio[(size_t)row * SEQ + tid + j * 256] = v[j] * inv;
}

// ---------------------------------------------------------------------------
// attn @ V:  O[b,q,h*64+n] = sum_k P[bh,q,k] * V[b,k,h*64+n].
// ---------------------------------------------------------------------------
__global__ __launch_bounds__(256) void attn_av_kernel(const float* __restrict__ P,
                                                      const float* __restrict__ V,
                                                      float* __restrict__ O) {
  __shared__ bf16_t Ps[64 * 32];
  __shared__ bf16_t VsT[64 * 32];
  const int tid = threadIdx.x, lane = tid & 31, w = tid >> 5;
  const int wq = (w >> 1) * 16, wn = (w & 1) * 32;
  const int q0 = blockIdx.x * 64;
  const int bh = blockIdx.z;
  const int b = bh / HEADS, h = bh % HEADS;
  const size_t pb = (size_t)bh * SEQ * SEQ;

  v8f acc0 = {0.f, 0.f, 0.f, 0.f, 0.f, 0.f, 0.f, 0.f};
  v8f acc1 = acc0;

  for (int kt = 0; kt < SEQ / 32; ++kt) {
    const int kb = kt * 32;
#pragma unroll
    for (int r = 0; r < 2; ++r) {
      int l4 = tid + r * 256;
      // P tile: rows q, contiguous k.
      int prow = l4 >> 3, pk = (l4 & 7) * 4;
      float4 fp = *(const float4*)&P[pb + (size_t)(q0 + prow) * SEQ + kb + pk];
      *(bf16x4*)&Ps[prow * 32 + pk] = pack4(fp.x, fp.y, fp.z, fp.w);
      // V tile: contiguous head-dim columns, scatter into transposed LDS.
      int kk = l4 >> 4, nq = (l4 & 15) * 4;
      float4 fv = *(const float4*)&V[(size_t)(b * SEQ + kb + kk) * DIM + h * HEAD_DIM + nq];
      VsT[(nq + 0) * 32 + kk] = (bf16_t)fv.x;
      VsT[(nq + 1) * 32 + kk] = (bf16_t)fv.y;
      VsT[(nq + 2) * 32 + kk] = (bf16_t)fv.z;
      VsT[(nq + 3) * 32 + kk] = (bf16_t)fv.w;
    }
    __syncthreads();
    v16bf ap = frag_a(&Ps[wq * 32], lane);
    v16bf b0 = frag_b_tr(&VsT[(wn) * 32], lane);
    v16bf b1 = frag_b_tr(&VsT[(wn + 16) * 32], lane);
    acc0 = wmma_bf16(ap, b0, acc0);
    acc1 = wmma_bf16(ap, b1, acc1);
    __syncthreads();
  }
  const int nlo = lane & 15, mh = lane >> 4;
#pragma unroll
  for (int r = 0; r < 8; ++r) {
    int q = q0 + wq + mh * 8 + r;
    O[(size_t)(b * SEQ + q) * DIM + h * HEAD_DIM + wn + nlo]      = acc0[r];
    O[(size_t)(b * SEQ + q) * DIM + h * HEAD_DIM + wn + 16 + nlo] = acc1[r];
  }
}

// ---------------------------------------------------------------------------
// Fused residual-add + LayerNorm (in-place capable).
// ---------------------------------------------------------------------------
__global__ __launch_bounds__(256) void add_ln_kernel(const float* __restrict__ X,
                                                     const float* __restrict__ R,
                                                     const float* __restrict__ g,
                                                     const float* __restrict__ bta,
                                                     float* __restrict__ Out) {
  __shared__ float red[256];
  const int row = blockIdx.x;
  const int tid = threadIdx.x;
  float v0 = X[(size_t)row * DIM + tid]       + R[(size_t)row * DIM + tid];
  float v1 = X[(size_t)row * DIM + tid + 256] + R[(size_t)row * DIM + tid + 256];
  red[tid] = v0 + v1;
  __syncthreads();
  for (int off = 128; off > 0; off >>= 1) {
    if (tid < off) red[tid] += red[tid + off];
    __syncthreads();
  }
  float mu = red[0] * (1.f / DIM);
  __syncthreads();
  float d0 = v0 - mu, d1 = v1 - mu;
  red[tid] = d0 * d0 + d1 * d1;
  __syncthreads();
  for (int off = 128; off > 0; off >>= 1) {
    if (tid < off) red[tid] += red[tid + off];
    __syncthreads();
  }
  float inv = rsqrtf(red[0] * (1.f / DIM) + LN_EPS);
  Out[(size_t)row * DIM + tid]       = d0 * inv * g[tid] + bta[tid];
  Out[(size_t)row * DIM + tid + 256] = d1 * inv * g[tid + 256] + bta[tid + 256];
}

// ---------------------------------------------------------------------------
// Embedding * sqrt(D) + sinusoidal positional encoding.
// ---------------------------------------------------------------------------
__global__ __launch_bounds__(256) void embed_kernel(const int* __restrict__ tok,
                                                    const float* __restrict__ emb,
                                                    float* __restrict__ X) {
  const int row = blockIdx.x;
  const int s   = row & (SEQ - 1);
  const int tid = threadIdx.x;
  const int t   = tok[row];
  const float sc = 22.62741699796952f;  // sqrt(512)
#pragma unroll
  for (int j = 0; j < 2; ++j) {
    int d = tid + j * 256;
    int i = d >> 1;
    float div = __expf((float)(2 * i) * (-9.210340371976184f / (float)DIM));
    float ang = (float)s * div;
    float pe = (d & 1) ? __cosf(ang) : __sinf(ang);
    X[(size_t)row * DIM + d] = emb[(size_t)t * DIM + d] * sc + pe;
  }
}

// ---------------------------------------------------------------------------
// Host-side orchestration
// ---------------------------------------------------------------------------
static void launch_gemm_w(const float* A, const float* Wf32, const float* bias,
                          float* C, int M, int N, int K, int has_bias, int relu,
                          bf16_t* wbf, hipStream_t s) {
  size_t nelem = (size_t)K * N;                       // all sizes % 1024 == 0
  cvt_bf16_kernel<<<(unsigned)(nelem / 1024), 256, 0, s>>>(Wf32, wbf);
  dim3 g(N / 128, M / 128), b(256);
  if (has_bias && relu)
    gemm_kernel<1, 1><<<g, b, 0, s>>>(A, wbf, bias, C, M, N, K);
  else if (has_bias)
    gemm_kernel<1, 0><<<g, b, 0, s>>>(A, wbf, bias, C, M, N, K);
  else
    gemm_kernel<0, 0><<<g, b, 0, s>>>(A, wbf, nullptr, C, M, N, K);
}

static void run_attention(const float* xq, const float* xkv,
                          const float* wq, const float* wk, const float* wv,
                          const float* wo, const int* key_tok, int causal,
                          float* qb, float* kb, float* vb, float* scores,
                          float* attn_cat, float* proj, bf16_t* wbf,
                          hipStream_t s) {
  launch_gemm_w(xq,  wq, nullptr, qb, TOK, DIM, DIM, 0, 0, wbf, s);
  launch_gemm_w(xkv, wk, nullptr, kb, TOK, DIM, DIM, 0, 0, wbf, s);
  launch_gemm_w(xkv, wv, nullptr, vb, TOK, DIM, DIM, 0, 0, wbf, s);
  dim3 gs(SEQ / 64, SEQ / 64, BATCH * HEADS);
  attn_scores_kernel<<<gs, 256, 0, s>>>(qb, kb, scores);
  softmax_kernel<<<BATCH * HEADS * SEQ, 256, 0, s>>>(scores, key_tok, causal);
  dim3 ga(SEQ / 64, 1, BATCH * HEADS);
  attn_av_kernel<<<ga, 256, 0, s>>>(scores, vb, attn_cat);
  launch_gemm_w(attn_cat, wo, nullptr, proj, TOK, DIM, DIM, 0, 0, wbf, s);
}

extern "C" void kernel_launch(void* const* d_in, const int* in_sizes, int n_in,
                              void* d_out, int out_size, void* d_ws, size_t ws_size,
                              hipStream_t stream) {
  (void)in_sizes; (void)n_in; (void)out_size; (void)ws_size;
  const int*   input  = (const int*)d_in[0];
  const int*   target = (const int*)d_in[1];
  const float* emb    = (const float*)d_in[2];
  const float* out_w  = (const float*)d_in[3];
  const float* out_b  = (const float*)d_in[4];
  const float* e_wq   = (const float*)d_in[5];
  const float* e_wk   = (const float*)d_in[6];
  const float* e_wv   = (const float*)d_in[7];
  const float* e_wo   = (const float*)d_in[8];
  const float* e_f1w  = (const float*)d_in[9];
  const float* e_f1b  = (const float*)d_in[10];
  const float* e_f2w  = (const float*)d_in[11];
  const float* e_f2b  = (const float*)d_in[12];
  const float* e_g1   = (const float*)d_in[13];
  const float* e_b1   = (const float*)d_in[14];
  const float* e_g2   = (const float*)d_in[15];
  const float* e_b2   = (const float*)d_in[16];
  const float* d_swq  = (const float*)d_in[17];
  const float* d_swk  = (const float*)d_in[18];
  const float* d_swv  = (const float*)d_in[19];
  const float* d_swo  = (const float*)d_in[20];
  const float* d_cwq  = (const float*)d_in[21];
  const float* d_cwk  = (const float*)d_in[22];
  const float* d_cwv  = (const float*)d_in[23];
  const float* d_cwo  = (const float*)d_in[24];
  const float* d_f1w  = (const float*)d_in[25];
  const float* d_f1b  = (const float*)d_in[26];
  const float* d_f2w  = (const float*)d_in[27];
  const float* d_f2b  = (const float*)d_in[28];
  const float* d_g1   = (const float*)d_in[29];
  const float* d_b1   = (const float*)d_in[30];
  const float* d_g2   = (const float*)d_in[31];
  const float* d_b2   = (const float*)d_in[32];
  const float* d_g3   = (const float*)d_in[33];
  const float* d_b3   = (const float*)d_in[34];

  // Workspace layout
  float* ws = (float*)d_ws;
  const size_t XD = (size_t)TOK * DIM;
  float* x_enc  = ws;
  float* x_dec  = x_enc + XD;
  float* qb     = x_dec + XD;
  float* kb     = qb + XD;
  float* vb     = kb + XD;
  float* ab     = vb + XD;
  float* pb     = ab + XD;
  float* ffb    = pb + XD;                              // TOK*FFDIM
  float* scores = ffb + (size_t)TOK * FFDIM;            // B*H*S*S
  bf16_t* wbf   = (bf16_t*)(scores + (size_t)BATCH * HEADS * SEQ * SEQ);
  // wbf holds the bf16 copy of the current weight (max DIM*VOCAB elements).

  embed_kernel<<<TOK, 256, 0, stream>>>(input,  emb, x_enc);
  embed_kernel<<<TOK, 256, 0, stream>>>(target, emb, x_dec);

  const size_t WSZ = (size_t)DIM * DIM;
  const size_t F1  = (size_t)DIM * FFDIM;
  const size_t F2  = (size_t)FFDIM * DIM;

  // ---------------- Encoder ----------------
  for (int l = 0; l < LAYERS; ++l) {
    run_attention(x_enc, x_enc, e_wq + l * WSZ, e_wk + l * WSZ, e_wv + l * WSZ,
                  e_wo + l * WSZ, input, 0, qb, kb, vb, scores, ab, pb, wbf, stream);
    add_ln_kernel<<<TOK, 256, 0, stream>>>(x_enc, pb, e_g1 + l * DIM, e_b1 + l * DIM, x_enc);
    launch_gemm_w(x_enc, e_f1w + l * F1, e_f1b + (size_t)l * FFDIM, ffb,
                  TOK, FFDIM, DIM, 1, 1, wbf, stream);
    launch_gemm_w(ffb, e_f2w + l * F2, e_f2b + (size_t)l * DIM, pb,
                  TOK, DIM, FFDIM, 1, 0, wbf, stream);
    add_ln_kernel<<<TOK, 256, 0, stream>>>(x_enc, pb, e_g2 + l * DIM, e_b2 + l * DIM, x_enc);
  }

  // ---------------- Decoder ----------------
  for (int l = 0; l < LAYERS; ++l) {
    run_attention(x_dec, x_dec, d_swq + l * WSZ, d_swk + l * WSZ, d_swv + l * WSZ,
                  d_swo + l * WSZ, target, 1, qb, kb, vb, scores, ab, pb, wbf, stream);
    add_ln_kernel<<<TOK, 256, 0, stream>>>(x_dec, pb, d_g1 + l * DIM, d_b1 + l * DIM, x_dec);
    run_attention(x_dec, x_enc, d_cwq + l * WSZ, d_cwk + l * WSZ, d_cwv + l * WSZ,
                  d_cwo + l * WSZ, input, 0, qb, kb, vb, scores, ab, pb, wbf, stream);
    add_ln_kernel<<<TOK, 256, 0, stream>>>(x_dec, pb, d_g2 + l * DIM, d_b2 + l * DIM, x_dec);
    launch_gemm_w(x_dec, d_f1w + l * F1, d_f1b + (size_t)l * FFDIM, ffb,
                  TOK, FFDIM, DIM, 1, 1, wbf, stream);
    launch_gemm_w(ffb, d_f2w + l * F2, d_f2b + (size_t)l * DIM, pb,
                  TOK, DIM, FFDIM, 1, 0, wbf, stream);
    add_ln_kernel<<<TOK, 256, 0, stream>>>(x_dec, pb, d_g3 + l * DIM, d_b3 + l * DIM, x_dec);
  }

  // Final vocab projection
  launch_gemm_w(x_dec, out_w, out_b, (float*)d_out, TOK, VOCAB, DIM, 1, 0, wbf, stream);
}
